// SetAbstraction_68229850464408
// MI455X (gfx1250) — matrix operand order repaired and checked by reading, not compiled, not run on hardware
//
#include <hip/hip_runtime.h>
#include <hip/hip_fp16.h>
#include <math.h>

typedef __attribute__((ext_vector_type(16))) _Float16 v16h;
typedef __attribute__((ext_vector_type(8)))  _Float16 v8h;
typedef __attribute__((ext_vector_type(4)))  _Float16 v4h;
typedef __attribute__((ext_vector_type(8)))  float    v8f;

#define N_PTS   32768
#define M_PTS   8192
#define N_EDGES (M_PTS * 64)
#define C_IN    32
#define C_OUT   64
#define KDIM    35          // C_IN + 3
#define KPAD    64          // padded K for WMMA (2 x K=32 steps)
#define NRINGS  4
#define NCOLS   (NRINGS * C_OUT)   // 256 combined (ring,out) columns
#define RADIUS  0.2f
#define ROWS_PER_BLOCK 32

// ---------------------------------------------------------------------------
// Pack h = concat(x, pos) -> f16, zero-padded to KPAD columns (row-major).
// A fragment chunks are contiguous 16B runs in this layout.
// ---------------------------------------------------------------------------
__global__ __launch_bounds__(256) void pack_h(const float* __restrict__ x,
                                              const float* __restrict__ pos,
                                              _Float16* __restrict__ h16) {
  int id = blockIdx.x * 256 + threadIdx.x;           // over N_PTS * KPAD
  if (id >= N_PTS * KPAD) return;
  int i = id & (KPAD - 1);
  int n = id >> 6;
  float v = 0.0f;
  if (i < C_IN)      v = x[n * C_IN + i];
  else if (i < KDIM) v = pos[n * 3 + (i - C_IN)];
  h16[id] = (_Float16)v;
}

// ---------------------------------------------------------------------------
// Pack W [4,64,35] directly into per-lane WMMA B-fragment order:
//   frag[(s*16 + tile)*32 + lane][j] , j = 0..15
//   element j of lane's v16h = B[K][col],
//   K = s*32 + (lane>>4)*8 + (j&7) + (j>>3)*16, col = tile*16 + (lane&15), col = ring*64+o.
// Each lane's fragment is 32 contiguous bytes -> 2x global_load_b128.
// ---------------------------------------------------------------------------
__global__ __launch_bounds__(256) void pack_w_frag(const float* __restrict__ W,
                                                   _Float16* __restrict__ w16f) {
  int id = blockIdx.x * 256 + threadIdx.x;           // over 2*16*32*16 = 16384
  if (id >= 2 * 16 * 32 * 16) return;
  int j    = id & 15;
  int lane = (id >> 4) & 31;
  int tile = (id >> 9) & 15;
  int s    = id >> 13;
  int kk   = s * 32 + (lane >> 4) * 8 + (j & 7) + (j >> 3) * 16;
  int col  = tile * 16 + (lane & 15);
  float v = 0.0f;
  if (kk < KDIM) {
    int ring = col >> 6;
    int o    = col & (C_OUT - 1);
    v = W[(ring * C_OUT + o) * KDIM + kk];
  }
  w16f[id] = (_Float16)v;
}

// ---------------------------------------------------------------------------
// Node-table GEMM, LDS-free WMMA:
//   Y[(n*64 + o)*4 + ring] = (W_ring . h[n])_o  as f16
// Block: 256 threads = 8 waves; tile = 32 rows x 256 cols.
// Per wave: 2 row subtiles x 2 col tiles x 2 K-steps = 8 v_wmma.
// All operand fragments arrive via 128-bit global loads (A streamed once,
// B fragments L2-resident). EXEC is all-ones throughout (WMMA requirement).
// ---------------------------------------------------------------------------
__global__ __launch_bounds__(256) void feat_gemm_wmma(const _Float16* __restrict__ A,
                                                      const _Float16* __restrict__ Bf,
                                                      _Float16* __restrict__ Y) {
  const int tid     = threadIdx.x;
  const int rowBase = blockIdx.x * ROWS_PER_BLOCK;
  const int lane    = tid & 31;
  const int wave    = tid >> 5;
  const int m       = lane & 15;
  const int half    = lane >> 4;

  const v8h*  pa0 = (const v8h*)(A + (size_t)(rowBase + m) * KPAD);
  const v8h*  pa1 = (const v8h*)(A + (size_t)(rowBase + 16 + m) * KPAD);
  const v16h* pb  = (const v16h*)Bf;

  v8f acc[2][2] = {{{}, {}}, {{}, {}}};

#pragma unroll
  for (int s = 0; s < 2; ++s) {               // K steps of 32
    // A fragments: two contiguous 16B chunks per (row subtile)
    v16h a[2];
    {
      v8h lo0 = pa0[s * 4 + half];
      v8h hi0 = pa0[s * 4 + half + 2];
      v8h lo1 = pa1[s * 4 + half];
      v8h hi1 = pa1[s * 4 + half + 2];
#pragma unroll
      for (int j = 0; j < 8; ++j) {
        a[0][j] = lo0[j];  a[0][8 + j] = hi0[j];
        a[1][j] = lo1[j];  a[1][8 + j] = hi1[j];
      }
    }
    // B fragments: pre-swizzled, one 32B contiguous load per (s, col tile)
    v16h b[2];
#pragma unroll
    for (int tt = 0; tt < 2; ++tt)
      b[tt] = pb[(size_t)(s * 16 + (wave * 2 + tt)) * 32 + lane];

#pragma unroll
    for (int rt = 0; rt < 2; ++rt)
#pragma unroll
      for (int tt = 0; tt < 2; ++tt)
        acc[rt][tt] = __builtin_amdgcn_wmma_f32_16x16x32_f16(
            false, a[rt], false, b[tt], (short)0, acc[rt][tt], false, false);
  }

  // D layout: VGPR r -> M = r + 8*half, N = lane%16.
  // Store to (node, o, ring)-ordered f16 table for the edge-gather pass.
#pragma unroll
  for (int rt = 0; rt < 2; ++rt) {
#pragma unroll
    for (int tt = 0; tt < 2; ++tt) {
      const int col  = (wave * 2 + tt) * 16 + m;
      const int ring = col >> 6;
      const int o    = col & (C_OUT - 1);
#pragma unroll
      for (int r = 0; r < 8; ++r) {
        int row = rowBase + rt * 16 + r + 8 * half;
        Y[((size_t)row * C_OUT + o) * NRINGS + ring] = (_Float16)acc[rt][tt][r];
      }
    }
  }
}

// ---------------------------------------------------------------------------
// Zero scratch accumulators
// ---------------------------------------------------------------------------
__global__ __launch_bounds__(256) void zero_f(float* __restrict__ p, int n) {
  int id = blockIdx.x * 256 + threadIdx.x;
  if (id < n) p[id] = 0.0f;
}

// ---------------------------------------------------------------------------
// Edge pass: per (edge, channel) thread loads its 4 ring values as ONE
// aligned 8-byte load (table is (src,o,ring)-ordered f16, L2-resident),
// applies the Gaussian ring basis + radius cutoff, atomically scatters.
// ---------------------------------------------------------------------------
__global__ __launch_bounds__(256) void edge_scatter(const float* __restrict__ pos,
                                                    const int*   __restrict__ cidx,
                                                    const int*   __restrict__ esrc,
                                                    const int*   __restrict__ edst,
                                                    const _Float16* __restrict__ Y,
                                                    float* __restrict__ agg,
                                                    float* __restrict__ cnt) {
  const int tid = threadIdx.x;
  const int e   = blockIdx.x * 4 + (tid >> 6);
  const int o   = tid & 63;

  const int src = esrc[e];
  const int dst = edst[e];
  const int ctr = cidx[dst];

  float dx = pos[src * 3 + 0] - pos[ctr * 3 + 0];
  float dy = pos[src * 3 + 1] - pos[ctr * 3 + 1];
  float dz = pos[src * 3 + 2] - pos[ctr * 3 + 2];
  float dist  = sqrtf(dx * dx + dy * dy + dz * dz + 1e-12f);
  float valid = (dist <= RADIUS) ? 1.0f : 0.0f;

  const float inv_sigma = (float)NRINGS / RADIUS;          // sigma = R/4
  const float cstep     = RADIUS / (float)(NRINGS - 1);    // linspace(0,R,4)

  v4h yv = *(const v4h*)(Y + ((size_t)src * C_OUT + o) * NRINGS);

  float msg = 0.0f;
#pragma unroll
  for (int k = 0; k < NRINGS; ++k) {
    float t = (dist - cstep * (float)k) * inv_sigma;
    float b = __expf(-0.5f * t * t) * valid;
    msg += b * (float)yv[k];
  }
  atomicAdd(&agg[dst * C_OUT + o], msg);
  if (o == 0) atomicAdd(&cnt[dst], valid);
}

// ---------------------------------------------------------------------------
// BN batch stats: one block per channel, reduce over M_PTS rows
// ---------------------------------------------------------------------------
__global__ __launch_bounds__(256) void bn_stats(const float* __restrict__ agg,
                                                const float* __restrict__ cnt,
                                                float* __restrict__ mean,
                                                float* __restrict__ var) {
  __shared__ float s1[256];
  __shared__ float s2[256];
  const int o = blockIdx.x;
  float a1 = 0.0f, a2 = 0.0f;
  for (int m = threadIdx.x; m < M_PTS; m += 256) {
    float a = agg[m * C_OUT + o] / fmaxf(cnt[m], 1.0f);
    a1 += a;
    a2 += a * a;
  }
  s1[threadIdx.x] = a1;
  s2[threadIdx.x] = a2;
  __syncthreads();
  for (int st = 128; st > 0; st >>= 1) {
    if (threadIdx.x < st) {
      s1[threadIdx.x] += s1[threadIdx.x + st];
      s2[threadIdx.x] += s2[threadIdx.x + st];
    }
    __syncthreads();
  }
  if (threadIdx.x == 0) {
    float mu = s1[0] / (float)M_PTS;
    mean[o] = mu;
    var[o]  = s2[0] / (float)M_PTS - mu * mu;
  }
}

// ---------------------------------------------------------------------------
// BN affine + field-norm nonlinearity. Fields of 4 channels map to 4
// consecutive lanes (wave32), so the field L2-norm is two xor-shuffles.
// ---------------------------------------------------------------------------
__global__ __launch_bounds__(256) void bn_normnl(const float* __restrict__ agg,
                                                 const float* __restrict__ cnt,
                                                 const float* __restrict__ mean,
                                                 const float* __restrict__ var,
                                                 const float* __restrict__ gamma,
                                                 const float* __restrict__ beta,
                                                 const float* __restrict__ nbias,
                                                 float* __restrict__ out) {
  const int gid = blockIdx.x * 256 + threadIdx.x;
  const int row = gid >> 6;
  const int o   = gid & 63;

  float a = agg[row * C_OUT + o] / fmaxf(cnt[row], 1.0f);
  float y = (a - mean[o]) * rsqrtf(var[o] + 1e-5f) * gamma[o] + beta[o];

  float sq = y * y;
  sq += __shfl_xor(sq, 1, 32);
  sq += __shfl_xor(sq, 2, 32);
  float n   = sqrtf(sq + 1e-12f);
  float z   = n + nbias[o >> 2];
  float elu = (z > 0.0f) ? z : (__expf(z) - 1.0f);
  out[gid] = y * (elu / n);
}

// ---------------------------------------------------------------------------
// Launch
// ---------------------------------------------------------------------------
extern "C" void kernel_launch(void* const* d_in, const int* in_sizes, int n_in,
                              void* d_out, int out_size, void* d_ws, size_t ws_size,
                              hipStream_t stream) {
  (void)in_sizes; (void)n_in; (void)out_size; (void)ws_size;

  const float* x     = (const float*)d_in[0];
  const float* pos   = (const float*)d_in[1];
  const int*   idx   = (const int*)d_in[2];
  const int*   esrc  = (const int*)d_in[3];
  const int*   edst  = (const int*)d_in[4];
  const float* W     = (const float*)d_in[5];
  const float* gamma = (const float*)d_in[6];
  const float* beta  = (const float*)d_in[7];
  const float* nbias = (const float*)d_in[8];
  float*       out   = (float*)d_out;

  // Workspace layout (~22 MB)
  char* ws = (char*)d_ws;
  size_t off = 0;
  _Float16* h16  = (_Float16*)(ws + off); off += (size_t)N_PTS * KPAD * 2;     // 4 MiB
  off = (off + 65535) & ~(size_t)65535;
  _Float16* w16f = (_Float16*)(ws + off); off += (size_t)2 * 16 * 32 * 16 * 2; // 32 KiB
  off = (off + 65535) & ~(size_t)65535;
  _Float16* Y    = (_Float16*)(ws + off); off += (size_t)N_PTS * NCOLS * 2;    // 16 MiB
  off = (off + 255) & ~(size_t)255;
  float* agg  = (float*)(ws + off); off += (size_t)M_PTS * C_OUT * 4;          // 2 MiB
  float* cnt  = (float*)(ws + off); off += (size_t)M_PTS * 4;                  // contiguous after agg
  float* mean = (float*)(ws + off); off += 256;
  float* var  = (float*)(ws + off); off += 256;

  pack_h<<<(N_PTS * KPAD + 255) / 256, 256, 0, stream>>>(x, pos, h16);
  pack_w_frag<<<(2 * 16 * 32 * 16 + 255) / 256, 256, 0, stream>>>(W, w16f);
  feat_gemm_wmma<<<N_PTS / ROWS_PER_BLOCK, 256, 0, stream>>>(h16, w16f, Y);

  // zero agg + cnt (contiguous)
  zero_f<<<(M_PTS * C_OUT + M_PTS + 255) / 256, 256, 0, stream>>>(agg, M_PTS * C_OUT + M_PTS);

  edge_scatter<<<N_EDGES / 4, 256, 0, stream>>>(pos, idx, esrc, edst, Y, agg, cnt);
  bn_stats<<<C_OUT, 256, 0, stream>>>(agg, cnt, mean, var);
  bn_normnl<<<(M_PTS * C_OUT) / 256, 256, 0, stream>>>(agg, cnt, mean, var,
                                                       gamma, beta, nbias, out);
}
